// SwinBlock_16475494547491
// MI455X (gfx1250) — compile-verified, matrix operand order checked by reading, not run on hardware
//
#include <hip/hip_runtime.h>

typedef unsigned short u16;
typedef __attribute__((ext_vector_type(16))) __bf16 v16bf;
typedef __attribute__((ext_vector_type(8)))  float  v8f;

#define BATCH 64
#define HH 56
#define CC 128
#define LTOK 3136                 // 56*56
#define MROWS 200704              // 64*3136
#define HID 384

union Frag {
    uint4 q[2];
    u16   s[16];
    v16bf v;
};

__device__ __forceinline__ u16 f2bf(float f) {
    unsigned int u = __float_as_uint(f);
    unsigned int r = u + 0x7FFFu + ((u >> 16) & 1u);   // round-to-nearest-even
    return (u16)(r >> 16);
}

// windowed row -> source/destination token row (shift by 3 with wrap), B=64,H=W=56,WS=7
__device__ __forceinline__ int window_map(int rw) {
    int b = rw / LTOK;  int t = rw - b * LTOK;
    int win = t / 49;   int pos = t - win * 49;
    int wh = win >> 3, ww = win & 7;
    int ih = pos / 7,  iw = pos - ih * 7;
    int hs = wh * 7 + ih, ws = ww * 7 + iw;
    int h = hs + 3; if (h >= HH) h -= HH;
    int w = ws + 3; if (w >= HH) w -= HH;
    return b * LTOK + h * HH + w;
}

// ---------------- weight fp32 -> bf16 ----------------
__global__ void cvt_bf16_kernel(const float* __restrict__ in, u16* __restrict__ out, int n) {
    int i = blockIdx.x * blockDim.x + threadIdx.x;
    if (i < n) out[i] = f2bf(in[i]);
}

// ---------------- LayerNorm (+optional shift/window gather), fp32 in -> bf16 out ----
// one wave per row, lane handles 4 channels (C=128)
__global__ __launch_bounds__(256) void ln_kernel(const float* __restrict__ x,
                                                 const float* __restrict__ g,
                                                 const float* __restrict__ be,
                                                 u16* __restrict__ out, int windowed) {
    int wave = threadIdx.x >> 5, lane = threadIdx.x & 31;
    int row = blockIdx.x * 8 + wave;
    if (row >= MROWS) return;
    int src = windowed ? window_map(row) : row;
    const float4 v  = *(const float4*)(x + (size_t)src * CC + lane * 4);
    float s = v.x + v.y + v.z + v.w;
    for (int m = 1; m < 32; m <<= 1) s += __shfl_xor(s, m, 32);
    float mu = s * (1.0f / 128.0f);
    float dx = v.x - mu, dy = v.y - mu, dz = v.z - mu, dw = v.w - mu;
    float q = dx * dx + dy * dy + dz * dz + dw * dw;
    for (int m = 1; m < 32; m <<= 1) q += __shfl_xor(q, m, 32);
    float rstd = rsqrtf(q * (1.0f / 128.0f) + 1e-5f);
    const float4 gg = *(const float4*)(g + lane * 4);
    const float4 bb = *(const float4*)(be + lane * 4);
    ushort4 o;
    o.x = f2bf(dx * rstd * gg.x + bb.x);
    o.y = f2bf(dy * rstd * gg.y + bb.y);
    o.z = f2bf(dz * rstd * gg.z + bb.z);
    o.w = f2bf(dw * rstd * gg.w + bb.w);
    *(ushort4*)(out + (size_t)row * CC + lane * 4) = o;
}

// load a 16x32 bf16 operand tile (A-layout) from row-major global memory
__device__ __forceinline__ v16bf load_tile_g(const u16* __restrict__ base, int row0, int ldk,
                                             int k0, int lane) {
    int lr = lane & 15, half = lane >> 4;
    const u16* p = base + (size_t)(row0 + lr) * ldk + k0 + half * 8;
    Frag f;
    f.q[0] = *(const uint4*)(p);
    f.q[1] = *(const uint4*)(p + 16);
    return f.v;
}

// ---------------- generic GEMM  out = act(A[M,K] @ W[N,K]^T + bias) --------------
// One wave computes a 64x64 output tile: 16 WMMAs per 32-wide k-step,
// 8 operand-tile loads per k-step (2.0 WMMA per tile load).
// EPI: 0 = bf16 store (qkv), 1 = exact-GELU + bf16 store (mlp1),
//      2 = fp32 resid add, identity rows (mlp2 -> d_out),
//      3 = fp32 resid add + un-window/un-shift scatter (proj -> h)
template <int EPI>
__global__ __launch_bounds__(256) void gemm_kernel(const u16* __restrict__ A,
                                                   const u16* __restrict__ W,
                                                   const float* __restrict__ bias,
                                                   void* __restrict__ out,
                                                   const float* __restrict__ resid,
                                                   int M, int N, int K, int n_tiles) {
    int lane = threadIdx.x & 31;
    int gw = blockIdx.x * 8 + (threadIdx.x >> 5);
    int mt = gw / n_tiles, nt = gw - mt * n_tiles;
    if (mt * 64 >= M) return;
    int m0 = mt * 64, n0 = nt * 64;

    v8f acc[4][4] = {};
    for (int k0 = 0; k0 < K; k0 += 32) {
        v16bf a[4], b[4];
#pragma unroll
        for (int i = 0; i < 4; i++) a[i] = load_tile_g(A, m0 + i * 16, K, k0, lane);
#pragma unroll
        for (int j = 0; j < 4; j++) b[j] = load_tile_g(W, n0 + j * 16, K, k0, lane);
#pragma unroll
        for (int i = 0; i < 4; i++)
#pragma unroll
            for (int j = 0; j < 4; j++)
                acc[i][j] = __builtin_amdgcn_wmma_f32_16x16x32_bf16(false, a[i], false, b[j],
                                                                    (short)0, acc[i][j], false, false);
    }

    int lr = lane & 15, half = lane >> 4;
    float bv[4];
#pragma unroll
    for (int j = 0; j < 4; j++) bv[j] = bias[n0 + j * 16 + lr];

#pragma unroll
    for (int i = 0; i < 4; i++) {
#pragma unroll
        for (int j = 0; j < 4; j++) {
            int ncol = n0 + j * 16 + lr;
#pragma unroll
            for (int r = 0; r < 8; r++) {
                int mrow = m0 + i * 16 + r + half * 8;
                float val = acc[i][j][r] + bv[j];
                if (EPI == 0) {
                    ((u16*)out)[(size_t)mrow * N + ncol] = f2bf(val);
                } else if (EPI == 1) {
                    val = 0.5f * val * (1.0f + erff(val * 0.70710678118654752f));
                    ((u16*)out)[(size_t)mrow * N + ncol] = f2bf(val);
                } else if (EPI == 2) {
                    ((float*)out)[(size_t)mrow * N + ncol] =
                        resid[(size_t)mrow * N + ncol] + val;
                } else {  // EPI == 3: scatter through inverse shifted-window map
                    int rt = window_map(mrow);
                    ((float*)out)[(size_t)rt * N + ncol] =
                        resid[(size_t)rt * N + ncol] + val;
                }
            }
        }
    }
}

// ---------------- windowed attention, one block (4 waves = 4 heads) per window ----
__global__ __launch_bounds__(128) void attn_kernel(const u16* __restrict__ qkv,
                                                   u16* __restrict__ o) {
    // sMem layout: [0..24575]  sQKV 64x384 bf16 (rows >=49 zeroed)
    //              [24576..32767] sVT 4x32x64 (v transposed per head)
    // sP (per wave 64x64) aliases the first 16384 entries after Q/K are consumed.
    __shared__ u16 sMem[32768];
    u16(*sQKV)[384] = (u16(*)[384])sMem;
    u16* sVTbase = sMem + 24576;

    int tid = threadIdx.x;
    int win = blockIdx.x;
    size_t gbase = (size_t)win * 49 * HID;

    for (int idx = tid; idx < 64 * 48; idx += 128) {       // 48 x uint4 per row
        int r = idx / 48, c8 = idx - (idx / 48) * 48;
        uint4 val = {0u, 0u, 0u, 0u};
        if (r < 49) val = *(const uint4*)(qkv + gbase + (size_t)r * HID + c8 * 8);
        *(uint4*)(&sQKV[r][c8 * 8]) = val;
    }
    __syncthreads();
    // transpose V per head: sVT[h][d][tok] = qkv[tok][256 + h*32 + d]
    for (int idx = tid; idx < 4 * 32 * 64; idx += 128) {
        int hh = idx >> 11, rem = idx & 2047;
        int dd = rem >> 6, tok = rem & 63;
        sVTbase[hh * 2048 + dd * 64 + tok] = sQKV[tok][256 + hh * 32 + dd];
    }
    __syncthreads();

    int wave = tid >> 5, lane = tid & 31;
    int lr = lane & 15, half = lane >> 4;
    int h = wave;

    // ---- S = Q Kh^T  (16 WMMAs, K=32=d in one step) ----
    v16bf Aq[4];
#pragma unroll
    for (int mtt = 0; mtt < 4; mtt++) {
        Frag f;
        const u16* p = &sQKV[mtt * 16 + lr][h * 32 + half * 8];
        f.q[0] = *(const uint4*)p;
        f.q[1] = *(const uint4*)(p + 16);
        Aq[mtt] = f.v;
    }
    v8f S[4][4] = {};
#pragma unroll
    for (int ntt = 0; ntt < 4; ntt++) {
        Frag f;
        const u16* p = &sQKV[ntt * 16 + lr][128 + h * 32 + half * 8];
        f.q[0] = *(const uint4*)p;
        f.q[1] = *(const uint4*)(p + 16);
        v16bf Bk = f.v;
#pragma unroll
        for (int mtt = 0; mtt < 4; mtt++)
            S[mtt][ntt] = __builtin_amdgcn_wmma_f32_16x16x32_bf16(false, Aq[mtt], false, Bk,
                                                                  (short)0, S[mtt][ntt], false, false);
    }

    // ---- scale + mask + softmax (rows striped: half0 -> M=r, half1 -> M=r+8) ----
    const float scale = 0.17677669529663687f;  // 1/sqrt(32)
#pragma unroll
    for (int ntt = 0; ntt < 4; ntt++) {
        int n = ntt * 16 + lr;
#pragma unroll
        for (int mtt = 0; mtt < 4; mtt++)
#pragma unroll
            for (int r = 0; r < 8; r++) {
                float v = S[mtt][ntt][r] * scale;
                S[mtt][ntt][r] = (n >= 49) ? -1e30f : v;
            }
    }
#pragma unroll
    for (int mtt = 0; mtt < 4; mtt++) {
#pragma unroll
        for (int r = 0; r < 8; r++) {
            float mx = -1e30f;
#pragma unroll
            for (int ntt = 0; ntt < 4; ntt++) mx = fmaxf(mx, S[mtt][ntt][r]);
            for (int d = 1; d < 16; d <<= 1) mx = fmaxf(mx, __shfl_xor(mx, d, 32));
            float sum = 0.0f;
#pragma unroll
            for (int ntt = 0; ntt < 4; ntt++) {
                float e = __expf(S[mtt][ntt][r] - mx);
                S[mtt][ntt][r] = e;
                sum += e;
            }
            for (int d = 1; d < 16; d <<= 1) sum += __shfl_xor(sum, d, 32);
            float inv = 1.0f / sum;
#pragma unroll
            for (int ntt = 0; ntt < 4; ntt++) S[mtt][ntt][r] *= inv;
        }
    }

    __syncthreads();                 // everyone finished reading Q/K from sMem
    u16* sP = sMem + h * 4096;       // wave-private 64x64 bf16, aliases dead Q/K
#pragma unroll
    for (int mtt = 0; mtt < 4; mtt++)
#pragma unroll
        for (int ntt = 0; ntt < 4; ntt++)
#pragma unroll
            for (int r = 0; r < 8; r++)
                sP[(mtt * 16 + r + half * 8) * 64 + ntt * 16 + lr] = f2bf(S[mtt][ntt][r]);
    __syncthreads();

    // ---- O = P @ Vh  (16 WMMAs: 4 M-tiles x 2 N-tiles x 2 K-steps) ----
    v8f O[4][2] = {};
#pragma unroll
    for (int kt = 0; kt < 2; kt++) {
        v16bf Bv[2];
#pragma unroll
        for (int j = 0; j < 2; j++) {
            Frag f;
            const u16* p = sVTbase + h * 2048 + (j * 16 + lr) * 64 + kt * 32 + half * 8;
            f.q[0] = *(const uint4*)p;
            f.q[1] = *(const uint4*)(p + 16);
            Bv[j] = f.v;
        }
#pragma unroll
        for (int mtt = 0; mtt < 4; mtt++) {
            Frag f;
            const u16* p = sP + (mtt * 16 + lr) * 64 + kt * 32 + half * 8;
            f.q[0] = *(const uint4*)p;
            f.q[1] = *(const uint4*)(p + 16);
            v16bf Ap = f.v;
#pragma unroll
            for (int j = 0; j < 2; j++)
                O[mtt][j] = __builtin_amdgcn_wmma_f32_16x16x32_bf16(false, Ap, false, Bv[j],
                                                                    (short)0, O[mtt][j], false, false);
        }
    }
#pragma unroll
    for (int mtt = 0; mtt < 4; mtt++)
#pragma unroll
        for (int j = 0; j < 2; j++)
#pragma unroll
            for (int r = 0; r < 8; r++) {
                int mrow = mtt * 16 + r + half * 8;
                if (mrow < 49) {
                    int col = h * 32 + j * 16 + lr;
                    o[(size_t)(win * 49 + mrow) * CC + col] = f2bf(O[mtt][j][r]);
                }
            }
}

// ------------------------------- launch ----------------------------------------
extern "C" void kernel_launch(void* const* d_in, const int* in_sizes, int n_in,
                              void* d_out, int out_size, void* d_ws, size_t ws_size,
                              hipStream_t stream) {
    const float* x     = (const float*)d_in[0];
    const float* g1    = (const float*)d_in[1];
    const float* be1   = (const float*)d_in[2];
    const float* Wqkv  = (const float*)d_in[3];
    const float* bqkv  = (const float*)d_in[4];
    const float* Wproj = (const float*)d_in[5];
    const float* bproj = (const float*)d_in[6];
    const float* g2    = (const float*)d_in[7];
    const float* be2   = (const float*)d_in[8];
    const float* Wm1   = (const float*)d_in[9];
    const float* bm1   = (const float*)d_in[10];
    const float* Wm2   = (const float*)d_in[11];
    const float* bm2   = (const float*)d_in[12];
    float* out = (float*)d_out;

    char* ws = (char*)d_ws;
    float* buf_h   = (float*)ws;                               // 200704*128 f32  = 102,760,448 B
    u16*   buf_hw  = (u16*)(ws + 102760448);                   // 200704*128 bf16 =  51,380,224 B
    u16*   buf_qkv = (u16*)(ws + 102760448 + 51380224);        // 200704*384 bf16 = 154,140,672 B (reused as MLP hidden)
    u16*   buf_o   = (u16*)(ws + 102760448 + 51380224 + 154140672);  // 51,380,224 B
    u16*   wbase   = (u16*)(ws + 359661568);
    u16* Wqkv_bf = wbase;            // 384*128
    u16* Wproj_bf = wbase + 49152;   // 128*128
    u16* Wm1_bf = wbase + 65536;     // 384*128
    u16* Wm2_bf = wbase + 114688;    // 128*384
    u16* buf_m = buf_qkv;            // alias: 200704*384 bf16

    // 0) weights to bf16
    cvt_bf16_kernel<<<(49152 + 255) / 256, 256, 0, stream>>>(Wqkv,  Wqkv_bf,  49152);
    cvt_bf16_kernel<<<(16384 + 255) / 256, 256, 0, stream>>>(Wproj, Wproj_bf, 16384);
    cvt_bf16_kernel<<<(49152 + 255) / 256, 256, 0, stream>>>(Wm1,   Wm1_bf,   49152);
    cvt_bf16_kernel<<<(49152 + 255) / 256, 256, 0, stream>>>(Wm2,   Wm2_bf,   49152);

    // 1) LN1 + shift + window partition -> bf16 (windowed row order)
    ln_kernel<<<MROWS / 8, 256, 0, stream>>>(x, g1, be1, buf_hw, 1);

    // 2) QKV GEMM: [200704,128] x [384,128]^T + bqkv -> bf16
    {
        int ntiles = HID / 64;                    // 6
        int waves = (MROWS / 64) * ntiles;        // 3136*6
        gemm_kernel<0><<<(waves + 7) / 8, 256, 0, stream>>>(buf_hw, Wqkv_bf, bqkv,
                                                            buf_qkv, nullptr, MROWS, HID, CC, ntiles);
    }

    // 3) windowed attention (4096 windows)
    attn_kernel<<<4096, 128, 0, stream>>>(buf_qkv, buf_o);

    // 4) proj GEMM + un-window/un-shift scatter + residual -> fp32 h
    {
        int ntiles = CC / 64;                     // 2
        int waves = (MROWS / 64) * ntiles;
        gemm_kernel<3><<<(waves + 7) / 8, 256, 0, stream>>>(buf_o, Wproj_bf, bproj,
                                                            buf_h, x, MROWS, CC, CC, ntiles);
    }

    // 5) LN2 -> bf16 (token order, reuses buf_hw)
    ln_kernel<<<MROWS / 8, 256, 0, stream>>>(buf_h, g2, be2, buf_hw, 0);

    // 6) MLP1 GEMM + exact GELU -> bf16 hidden (aliases dead qkv buffer)
    {
        int ntiles = HID / 64;
        int waves = (MROWS / 64) * ntiles;
        gemm_kernel<1><<<(waves + 7) / 8, 256, 0, stream>>>(buf_hw, Wm1_bf, bm1,
                                                            buf_m, nullptr, MROWS, HID, CC, ntiles);
    }

    // 7) MLP2 GEMM + residual add -> fp32 output
    {
        int ntiles = CC / 64;
        int waves = (MROWS / 64) * ntiles;
        gemm_kernel<2><<<(waves + 7) / 8, 256, 0, stream>>>(buf_m, Wm2_bf, bm2,
                                                            out, buf_h, MROWS, CC, HID, ntiles);
    }
}
